// VMPOModel_19129784336415
// MI455X (gfx1250) — matrix operand order repaired and checked by reading, not compiled, not run on hardware
//
#include <hip/hip_runtime.h>
#include <hip/hip_bf16.h>
#include <math.h>

// ---- problem dims ----
#define B_    2048
#define S_    64
#define D_    6
#define H_    4
#define DM_   64
#define SS_   150
#define SSP   160      // SS padded to multiple of 32 for WMMA K
#define A_    7
#define HID_  256
#define FF_   256
#define DH_   16
#define KV_   128      // 2*S

// ---- flat output offsets (mu, std, seq, mem_out, new_length) ----
#define OUT_MU  0
#define OUT_STD (B_*A_)
#define OUT_SEQ (2*B_*A_)
#define OUT_MEM (OUT_SEQ + (size_t)S_*B_*SS_)
#define OUT_LEN (OUT_MEM + (size_t)D_*B_*S_*DM_)

typedef __attribute__((ext_vector_type(16))) __bf16 v16bf;
typedef __attribute__((ext_vector_type(8)))  float  v8f;

// ---- WMMA fragment loaders (wave32; contiguous k order, same in A and B) ----
// A (16x32 bf16), row-major source: lane L holds row M = L%16;
// half i holds K = i + 16*(L/16)  -> one contiguous 16-half chunk.
__device__ inline v16bf load_a_frag(const __bf16* A, int m0, int k0, int stride, int lane) {
  v16bf a;
  const __bf16* row = A + (m0 + (lane & 15)) * stride + k0 + ((lane >> 4) << 4);
#pragma unroll
  for (int i = 0; i < 16; ++i)
    a[i] = row[i];
  return a;
}
// B fragment from *swizzled* storage: B is stored in 32(K)x16(N) tiles of 512
// halves, tile id = (k/32)*(N/16) + (n/16); within a tile, lane L's 16 halves
// are contiguous at L*16 (L = (n%16) + 16*((k%32)/16), half index = k%16).
__device__ inline v16bf load_b_frag_sw(const __bf16* Bsw, int k0, int n0, int Ntot, int lane) {
  const __bf16* p = Bsw + (((k0 >> 5) * (Ntot >> 4) + (n0 >> 4)) << 9) + (lane << 4);
  v16bf b;
#pragma unroll
  for (int i = 0; i < 16; ++i) b[i] = p[i];
  return b;
}
// swizzled-store index for B element (k,n) of a KxN matrix
__device__ inline int b_sw_idx(int k, int n, int N) {
  return (((k >> 5) * (N >> 4) + (n >> 4)) << 9) +
         (((n & 15) + (((k >> 4) & 1) << 4)) << 4) + (k & 15);
}
// stage a KxN f32 global matrix into swizzled bf16 LDS
__device__ inline void stage_b_sw(__bf16* dst, const float* __restrict__ src,
                                  int K, int N, int tid) {
  for (int idx = tid; idx < K * N; idx += 256) {
    int k = idx / N, n = idx - k * N;
    dst[b_sw_idx(k, n, N)] = (__bf16)src[idx];
  }
}

// Generic LDS->LDS GEMM over 16x16 C tiles, 8 waves round-robin on tiles.
// A row-major (stride sA, row offset aRow0); B swizzled with total width N.
template <class StoreF>
__device__ inline void wmma_gemm(const __bf16* A, int aRow0, int sA,
                                 const __bf16* Bsw,
                                 int M, int N, int K, int tid, StoreF&& st) {
  const int wave = tid >> 5, lane = tid & 31;
  const int tN = N >> 4;
  const int tiles = (M >> 4) * tN;
  for (int t = wave; t < tiles; t += 8) {
    int tm = t / tN, tn = t - tm * tN;
    v8f c = {};
    for (int k0 = 0; k0 < K; k0 += 32) {
      v16bf a = load_a_frag(A, aRow0 + tm * 16, k0, sA, lane);
      v16bf b = load_b_frag_sw(Bsw, k0, tn * 16, N, lane);
      c = __builtin_amdgcn_wmma_f32_16x16x32_bf16(false, a, false, b, (short)0, c,
                                                  false, false);
    }
    int m0 = tm * 16 + ((lane >> 4) << 3);
    int n  = tn * 16 + (lane & 15);
#pragma unroll
    for (int r = 0; r < 8; ++r) st(m0 + r, n, c[r]);
  }
}

// ---------- kernel 1: copy sequence buffer to output (vectorized b128) ----------
__global__ void seq_copy_kernel(const float4* __restrict__ src, float4* __restrict__ dst, int n4) {
  for (int i = blockIdx.x * blockDim.x + threadIdx.x; i < n4; i += gridDim.x * blockDim.x)
    dst[i] = src[i];
}

// ---------- kernel 2: LayerNorm(observation) scattered into seq at length[b] ----------
__global__ __launch_bounds__(64) void obs_scatter_kernel(
    const float* __restrict__ obs, const int* __restrict__ length,
    const float* __restrict__ g, const float* __restrict__ beta,
    float* __restrict__ seqOut) {
  __shared__ float s1[64], s2[64], stats[2];
  int b = blockIdx.x, t = threadIdx.x;
  float s = 0.f, ss = 0.f;
  for (int e = t; e < SS_; e += 64) {
    float v = obs[b * SS_ + e];
    s += v; ss += v * v;
  }
  s1[t] = s; s2[t] = ss;
  __syncthreads();
  if (t == 0) {
    float a = 0.f, c = 0.f;
    for (int i = 0; i < 64; ++i) { a += s1[i]; c += s2[i]; }
    float m = a / SS_;
    float var = c / SS_ - m * m;
    stats[0] = m; stats[1] = rsqrtf(var + 1e-5f);
  }
  __syncthreads();
  int len = length[b];
  int lidx = len < 0 ? 0 : (len > S_ - 1 ? S_ - 1 : len);
  float m = stats[0], rs = stats[1];
  for (int e = t; e < SS_; e += 64)
    seqOut[((size_t)lidx * B_ + b) * SS_ + e] = (obs[b * SS_ + e] - m) * rs * g[e] + beta[e];
}

// ---------- kernel 3: fused transformer + policy head, one block per batch ----------
__global__ __launch_bounds__(256) void vmpo_main_kernel(
    const float* __restrict__ seqUpd, const float* __restrict__ memIn,
    const int* __restrict__ length, const float* __restrict__ W_emb,
    const float* __restrict__ ln1_g, const float* __restrict__ ln1_b,
    const float* __restrict__ ln2_g, const float* __restrict__ ln2_b,
    const float* __restrict__ Wq, const float* __restrict__ Wk,
    const float* __restrict__ Wv, const float* __restrict__ Wo,
    const float* __restrict__ W1, const float* __restrict__ b1,
    const float* __restrict__ W2, const float* __restrict__ b2,
    const float* __restrict__ Wp1, const float* __restrict__ bp1,
    const float* __restrict__ Wp2, const float* __restrict__ bp2,
    float* __restrict__ out) {
  extern __shared__ char smem[];
  float*  x    = (float*)(smem);                    // 64x64 f32 residual      16KB
  float*  att  = (float*)(smem + 16384);            // 64x64 f32 attn out      16KB (reused: ph)
  float*  scf  = (float*)(smem + 32768);            // 64x128 f32 scores       32KB
  __bf16* midb = (__bf16*)(smem + 32768);           //   overlay: 64x256 bf16 FFN mid
  __bf16* wbuf = (__bf16*)(smem + 65536);           // swizzled weights        32KB
  __bf16* u1   = (__bf16*)(smem + 98304);           // 64x160 embed A / 64x128 attn bf16  20KB
  __bf16* kv   = (__bf16*)(smem + 118784);          // 128x64 bf16 (mem|h)     16KB
  __bf16* qb   = (__bf16*)(smem + 135168);          // 4x64x32 bf16 (q, K-pad) 16KB
  __bf16* ktb  = (__bf16*)(smem + 151552);          // 4 heads swizzled k^T    32KB
  __bf16* vbuf = (__bf16*)(smem + 184320);          // 4 heads swizzled v      16KB  => 196KB
  float*  ph   = att;                               // 256 f32 policy hidden (reuse)

  const int b = blockIdx.x, tid = threadIdx.x;
  const int len = length[b];
  const int lidx = len < 0 ? 0 : (len > S_ - 1 ? S_ - 1 : len);
  const int newlen = (len + 1) & (S_ - 1);
  const bool reset = (newlen == 0);

  // ---- embedding: x = seq_b (64x150) @ W_emb (150x64), K padded to 160 ----
  for (int idx = tid; idx < S_ * SSP; idx += 256) {
    int s = idx / SSP, e = idx - s * SSP;
    float v = (e < SS_) ? seqUpd[((size_t)s * B_ + b) * SS_ + e] : 0.f;
    u1[idx] = (__bf16)v;
  }
  for (int idx = tid; idx < SSP * DM_; idx += 256) {
    int k = idx >> 6, n = idx & 63;
    float v = (k < SS_) ? W_emb[k * DM_ + n] : 0.f;
    wbuf[b_sw_idx(k, n, DM_)] = (__bf16)v;
  }
  for (int idx = tid; idx < H_ * S_ * 32; idx += 256)  qb[idx]  = (__bf16)0.f;  // zero K-pad
  for (int idx = tid; idx < H_ * 32 * KV_; idx += 256) ktb[idx] = (__bf16)0.f;
  __syncthreads();
  wmma_gemm(u1, 0, SSP, wbuf, S_, DM_, SSP, tid,
            [&](int m, int n, float v) { x[m * DM_ + n] = v; });
  __syncthreads();

  const float scale = 0.25f;  // 1/sqrt(DH)
  for (int l = 0; l < D_; ++l) {
    const float* mptr  = memIn + ((size_t)l * B_ + b) * (S_ * DM_);
    float*       moptr = out + OUT_MEM + ((size_t)l * B_ + b) * (S_ * DM_);
    // fused memory reset/copy (b128) + stage memory rows into kv[0..63]
    for (int idx = tid; idx < (S_ * DM_) / 4; idx += 256) {
      float4 mv = ((const float4*)mptr)[idx];
      ((float4*)moptr)[idx] = reset ? ((const float4*)x)[idx] : mv;
      __bf16* kd = kv + idx * 4;
      kd[0] = (__bf16)mv.x; kd[1] = (__bf16)mv.y;
      kd[2] = (__bf16)mv.z; kd[3] = (__bf16)mv.w;
    }
    __syncthreads();
    // LN1(x) -> kv rows 64..127
    if (tid < S_) {
      const float* xr = x + tid * DM_;
      float m = 0.f;
      for (int d = 0; d < DM_; ++d) m += xr[d];
      m *= (1.f / DM_);
      float v2 = 0.f;
      for (int d = 0; d < DM_; ++d) { float t = xr[d] - m; v2 += t * t; }
      float rs = rsqrtf(v2 * (1.f / DM_) + 1e-5f);
      for (int d = 0; d < DM_; ++d)
        kv[(S_ + tid) * DM_ + d] =
            (__bf16)((xr[d] - m) * rs * ln1_g[l * DM_ + d] + ln1_b[l * DM_ + d]);
    }
    stage_b_sw(wbuf, Wq + (size_t)l * DM_ * DM_, DM_, DM_, tid);
    __syncthreads();
    // q = h @ Wq -> per-head K-padded row-major layout qb[h][row][0..15]
    wmma_gemm(kv, S_, DM_, wbuf, S_, DM_, DM_, tid,
              [&](int m, int n, float v) {
                qb[((n >> 4) * S_ + m) * 32 + (n & 15)] = (__bf16)v;
              });
    __syncthreads();
    stage_b_sw(wbuf, Wk + (size_t)l * DM_ * DM_, DM_, DM_, tid);
    __syncthreads();
    // k = kv_in @ Wk -> per-head swizzled B (K=head dim padded to 32, N=128 pos)
    wmma_gemm(kv, 0, DM_, wbuf, KV_, DM_, DM_, tid,
              [&](int m, int n, float v) {
                // head = n>>4, b-element (k = n&15, col = m), Ntot = 128
                ktb[((n >> 4) << 12) + ((m >> 4) << 9) + ((m & 15) << 4) + (n & 15)] =
                    (__bf16)v;
              });
    __syncthreads();
    stage_b_sw(wbuf, Wv + (size_t)l * DM_ * DM_, DM_, DM_, tid);
    __syncthreads();
    // v = kv_in @ Wv -> per-head swizzled B (K=128 pos, N=16 dims)
    wmma_gemm(kv, 0, DM_, wbuf, KV_, DM_, DM_, tid,
              [&](int m, int n, float v) {
                vbuf[((n >> 4) << 11) + ((m >> 5) << 9) +
                     (((n & 15) + (((m >> 4) & 1) << 4)) << 4) + (m & 15)] = (__bf16)v;
              });
    __syncthreads();

    // prefetch next layer's weights while attention runs (global_prefetch_b8)
    if (l + 1 < D_) {
      int line = (tid & 63) << 6;                 // 64-float (256B) granules
      __builtin_prefetch(Wq + (size_t)(l + 1) * 4096 + line, 0, 1);
      __builtin_prefetch(Wk + (size_t)(l + 1) * 4096 + line, 0, 1);
      __builtin_prefetch(Wv + (size_t)(l + 1) * 4096 + line, 0, 1);
      __builtin_prefetch(Wo + (size_t)(l + 1) * 4096 + line, 0, 1);
      int line2 = tid << 6;
      __builtin_prefetch(W1 + (size_t)(l + 1) * 16384 + line2, 0, 1);
      __builtin_prefetch(W2 + (size_t)(l + 1) * 16384 + line2, 0, 1);
    }

    // ---- attention, head by head ----
    for (int hh = 0; hh < H_; ++hh) {
      // scores (64x128) = q_h (64x16 pad 32) @ k_h^T (32x128, swizzled)
      wmma_gemm(qb + hh * S_ * 32, 0, 32, ktb + (hh << 12), S_, KV_, 32, tid,
                [&](int m, int n, float v) { scf[m * KV_ + n] = v * scale; });
      __syncthreads();
      // masked softmax, one thread per query row; write attn as bf16 into u1
      if (tid < S_) {
        float* sr = scf + tid * KV_;
        float mx = -1e30f;
        for (int n = 0; n < KV_; ++n) {
          bool valid = (n < S_) || (n - S_ <= tid);
          if (valid && sr[n] > mx) mx = sr[n];
        }
        float sum = 0.f;
        for (int n = 0; n < KV_; ++n) {
          bool valid = (n < S_) || (n - S_ <= tid);
          float e = valid ? __expf(sr[n] - mx) : 0.f;
          sr[n] = e; sum += e;
        }
        float inv = 1.f / sum;
        for (int n = 0; n < KV_; ++n) u1[tid * KV_ + n] = (__bf16)(sr[n] * inv);
      }
      __syncthreads();
      // out_h (64x16) = attn (64x128) @ v_h (128x16, swizzled)
      wmma_gemm(u1, 0, KV_, vbuf + (hh << 11), S_, DH_, KV_, tid,
                [&](int m, int n, float v) { att[m * DM_ + hh * DH_ + n] = v; });
      __syncthreads();
    }

    // ---- x += att @ Wo ----
    for (int idx = tid; idx < S_ * DM_; idx += 256) u1[idx] = (__bf16)att[idx];
    stage_b_sw(wbuf, Wo + (size_t)l * DM_ * DM_, DM_, DM_, tid);
    __syncthreads();
    wmma_gemm(u1, 0, DM_, wbuf, S_, DM_, DM_, tid,
              [&](int m, int n, float v) { x[m * DM_ + n] += v; });
    __syncthreads();

    // ---- FFN: x += relu(LN2(x) @ W1 + b1) @ W2 + b2 ----
    if (tid < S_) {
      const float* xr = x + tid * DM_;
      float m = 0.f;
      for (int d = 0; d < DM_; ++d) m += xr[d];
      m *= (1.f / DM_);
      float v2 = 0.f;
      for (int d = 0; d < DM_; ++d) { float t = xr[d] - m; v2 += t * t; }
      float rs = rsqrtf(v2 * (1.f / DM_) + 1e-5f);
      for (int d = 0; d < DM_; ++d)
        kv[(S_ + tid) * DM_ + d] =
            (__bf16)((xr[d] - m) * rs * ln2_g[l * DM_ + d] + ln2_b[l * DM_ + d]);
    }
    stage_b_sw(wbuf, W1 + (size_t)l * DM_ * FF_, DM_, FF_, tid);
    __syncthreads();
    wmma_gemm(kv, S_, DM_, wbuf, S_, FF_, DM_, tid,
              [&](int m, int n, float v) {
                float t = v + b1[l * FF_ + n];
                midb[m * FF_ + n] = (__bf16)(t > 0.f ? t : 0.f);
              });
    __syncthreads();
    stage_b_sw(wbuf, W2 + (size_t)l * FF_ * DM_, FF_, DM_, tid);
    __syncthreads();
    wmma_gemm(midb, 0, FF_, wbuf, S_, DM_, FF_, tid,
              [&](int m, int n, float v) { x[m * DM_ + n] += v + b2[l * DM_ + n]; });
    __syncthreads();
  }

  // ---- policy head on gathered row x[lidx] ----
  {
    const float* xr = x + lidx * DM_;
    float acc = bp1[tid];
    for (int d = 0; d < DM_; ++d) acc += xr[d] * Wp1[d * HID_ + tid];
    ph[tid] = acc > 0.f ? acc : 0.f;
  }
  __syncthreads();
  if (tid < 2 * A_) {
    float acc = bp2[tid];
    for (int j = 0; j < HID_; ++j) acc += ph[j] * Wp2[j * (2 * A_) + tid];
    if (tid < A_) {
      float sp = (acc > 20.f) ? acc : log1pf(__expf(acc));
      out[OUT_STD + (size_t)b * A_ + tid] = sp;      // std = softplus(pol[:, :A])
    } else {
      out[OUT_MU + (size_t)b * A_ + (tid - A_)] = tanhf(acc);  // mu = tanh(pol[:, A:])
    }
  }
  if (tid == 0) out[OUT_LEN + b] = (float)newlen;
}

extern "C" void kernel_launch(void* const* d_in, const int* in_sizes, int n_in,
                              void* d_out, int out_size, void* d_ws, size_t ws_size,
                              hipStream_t stream) {
  const float* obs    = (const float*)d_in[0];
  const float* seq    = (const float*)d_in[1];
  const float* mem    = (const float*)d_in[2];
  const int*   length = (const int*)d_in[3];
  const float* ln_g   = (const float*)d_in[4];
  const float* ln_b   = (const float*)d_in[5];
  const float* W_emb  = (const float*)d_in[6];
  const float* ln1_g  = (const float*)d_in[7];
  const float* ln1_b  = (const float*)d_in[8];
  const float* ln2_g  = (const float*)d_in[9];
  const float* ln2_b  = (const float*)d_in[10];
  const float* Wq     = (const float*)d_in[11];
  const float* Wk     = (const float*)d_in[12];
  const float* Wv     = (const float*)d_in[13];
  const float* Wo     = (const float*)d_in[14];
  const float* W1     = (const float*)d_in[15];
  const float* b1     = (const float*)d_in[16];
  const float* W2     = (const float*)d_in[17];
  const float* b2     = (const float*)d_in[18];
  const float* Wp1    = (const float*)d_in[19];
  const float* bp1    = (const float*)d_in[20];
  const float* Wp2    = (const float*)d_in[21];
  const float* bp2    = (const float*)d_in[22];
  float* out = (float*)d_out;
  float* seqOut = out + OUT_SEQ;

  const int n4 = (S_ * B_ * SS_) / 4;
  seq_copy_kernel<<<4096, 256, 0, stream>>>((const float4*)seq, (float4*)seqOut, n4);
  obs_scatter_kernel<<<B_, 64, 0, stream>>>(obs, length, ln_g, ln_b, seqOut);
  const size_t shmem = 200704;  // 196KB dynamic LDS (carved above)
  vmpo_main_kernel<<<B_, 256, shmem, stream>>>(
      seqOut, mem, length, W_emb, ln1_g, ln1_b, ln2_g, ln2_b,
      Wq, Wk, Wv, Wo, W1, b1, W2, b2, Wp1, bp1, Wp2, bp2, out);
}
